// CapsuleNetwork_17497696764606
// MI455X (gfx1250) — matrix operand order repaired and checked by reading, not compiled, not run on hardware
//
#include <hip/hip_runtime.h>
#include <hip/hip_bf16.h>

#define HIDDEN 128
#define NUM_INTEREST 8
#define SEQ 200
#define BATCH 512
#define ITER_TIMES 3

#define S_PAD   208   // SEQ padded to multiple of 16 (M tiles for logit update GEMM)
#define S_PAD32 224   // SEQ padded to multiple of 32 (K dim for u = c @ item)
#define KPAD    16    // capsule count padded to 16 (M dim for u GEMM)

typedef __bf16 bf16_t;
typedef __attribute__((ext_vector_type(16))) __bf16 v16bf;
typedef __attribute__((ext_vector_type(8)))  __bf16 v8bf;
typedef __attribute__((ext_vector_type(8)))  float  v8f;

__device__ __forceinline__ bf16_t f2bf(float f) { return (bf16_t)f; }

// A-matrix (16x32 bf16) fragment per ISA 7.12.2:
// lane L (L<16): row M=L; elems 0..7 <- K=0..7, elems 8..15 <- K=16..23
// lane L (>=16): row M=L-16; elems 0..7 <- K=8..15, elems 8..15 <- K=24..31
// rowptr points at row M, K-base already added; half = (lane>=16)?8:0
__device__ __forceinline__ v16bf load_frag_A(const bf16_t* rowptr, int half) {
    v8bf lo = *(const v8bf*)(rowptr + half);
    v8bf hi = *(const v8bf*)(rowptr + half + 16);
    v16bf a;
#pragma unroll
    for (int i = 0; i < 8; ++i) { a[i] = lo[i]; a[8 + i] = hi[i]; }
    return a;
}

__global__ __launch_bounds__(256)
void capsule_routing_kernel(const float* __restrict__ item_g,
                            const int*   __restrict__ mask_g,
                            const float* __restrict__ W_g,
                            const float* __restrict__ b_lin_g,
                            float* __restrict__ out_g)
{
    // LDS (~140 KB total, < 320 KB/WGP)
    __shared__ __align__(64) bf16_t item_bf[S_PAD][HIDDEN];    // row-major item (A for logit-update GEMM)
    __shared__ __align__(64) bf16_t item_t [HIDDEN][S_PAD32];  // transposed item (B for u GEMM)
    __shared__ __align__(64) bf16_t c_bf   [KPAD][S_PAD32];    // routing weights (A for u GEMM), zero padded
    __shared__ __align__(64) bf16_t v_bf   [KPAD][HIDDEN];     // v[k] = W_k^T @ interest_k (B for logit-update)
    __shared__ __align__(16) float  b_logit[NUM_INTEREST][S_PAD];
    __shared__ __align__(16) float  u_l    [KPAD][HIDDEN];
    __shared__ __align__(16) float  inter_l[NUM_INTEREST][HIDDEN];
    __shared__ float csum_l[NUM_INTEREST];
    __shared__ float beta_l[NUM_INTEREST];
    __shared__ float scale_l[NUM_INTEREST];
    __shared__ int   mask_l[S_PAD];

    const int b      = blockIdx.x;
    const int tid    = threadIdx.x;
    const int lane   = tid & 31;
    const int wv     = tid >> 5;           // 8 wave32 waves
    const int half   = (lane >= 16) ? 8 : 0;
    const int col    = lane & 15;          // N index / row selector
    const int rowoff = (lane < 16) ? 0 : 8;
    const int boff   = (lane < 16) ? 0 : 16;

    // ---------------- init ----------------
    for (int i = tid; i < NUM_INTEREST * S_PAD; i += 256)
        (&b_logit[0][0])[i] = 0.0f;
    for (int i = tid; i < KPAD * S_PAD32; i += 256)
        (&c_bf[0][0])[i] = (bf16_t)0.0f;
    for (int i = tid; i < KPAD * HIDDEN; i += 256)
        (&v_bf[0][0])[i] = (bf16_t)0.0f;
    for (int s = tid; s < S_PAD; s += 256)
        mask_l[s] = (s < SEQ) ? mask_g[(size_t)b * SEQ + s] : 0;

    // pull W into cache hierarchy (global_prefetch_b8); W stays L2-resident
    __builtin_prefetch(W_g + (size_t)tid * 512, 0, 3);

    const float* itemb = item_g + (size_t)b * SEQ * HIDDEN;
    for (int i = tid; i < SEQ * HIDDEN; i += 256) {
        int s = i >> 7, d = i & 127;
        bf16_t v = f2bf(itemb[i]);
        item_bf[s][d] = v;
        item_t[d][s]  = v;
    }
    for (int i = tid; i < (S_PAD - SEQ) * HIDDEN; i += 256) {   // pad rows of item_bf
        int s = SEQ + (i >> 7), d = i & 127;
        item_bf[s][d] = (bf16_t)0.0f;
    }
    for (int i = tid; i < HIDDEN * (S_PAD32 - SEQ); i += 256) { // pad cols of item_t
        int d = i / (S_PAD32 - SEQ);
        int s = SEQ + (i % (S_PAD32 - SEQ));
        item_t[d][s] = (bf16_t)0.0f;
    }
    __syncthreads();

    // ---------------- dynamic routing ----------------
    for (int it = 0; it < ITER_TIMES; ++it) {
        // A: mask + softmax over capsules (per sequence position)
        if (tid < SEQ) {
            int s = tid;
            bool msk0 = (mask_l[s] == 0);
            float vals[NUM_INTEREST];
            float mx = -3.4e38f;
#pragma unroll
            for (int k = 0; k < NUM_INTEREST; ++k) {
                float v = msk0 ? -10000.0f : b_logit[k][s];
                if (msk0) b_logit[k][s] = -10000.0f;
                vals[k] = v;
                mx = fmaxf(mx, v);
            }
            float ssum = 0.0f;
#pragma unroll
            for (int k = 0; k < NUM_INTEREST; ++k) { vals[k] = expf(vals[k] - mx); ssum += vals[k]; }
            float inv = 1.0f / ssum;
#pragma unroll
            for (int k = 0; k < NUM_INTEREST; ++k)
                c_bf[k][s] = f2bf(vals[k] * inv);
        }
        __syncthreads();

        // B: u = c @ item  (16pad x 224pad @ 224pad x 128), WMMA bf16
        //    wave wv owns d-tile wv (8 waves x 16 cols = 128)
        {
            const int dtile = wv * 16;
            v8f acc = {0.f,0.f,0.f,0.f,0.f,0.f,0.f,0.f};
#pragma unroll
            for (int ks = 0; ks < S_PAD32; ks += 32) {
                v16bf a  = load_frag_A(&c_bf[col][ks], half);
                v16bf bb = load_frag_A(&item_t[dtile + col][ks], half);
                acc = __builtin_amdgcn_wmma_f32_16x16x32_bf16(false, a, false, bb,
                                                              (short)0, acc, false, false);
            }
#pragma unroll
            for (int r = 0; r < 8; ++r)
                u_l[r + rowoff][dtile + col] = acc[r];
        }
        __syncthreads();

        // C1: column sums of c (for bias term)
        if (tid < NUM_INTEREST) {
            float s = 0.0f;
            for (int j = 0; j < SEQ; ++j) s += (float)c_bf[tid][j];
            csum_l[tid] = s;
        }
        __syncthreads();

        // C2: interest_pre[k,d'] = W_k[d',:] . u[k,:] + csum[k]*bias  (fp32, W from L2)
        for (int j = 0; j < 4; ++j) {
            int o = tid + j * 256;             // o = k*128 + d'
            int k = o >> 7;
            const float* wrow = W_g + (size_t)o * HIDDEN;
            float acc = csum_l[k] * b_lin_g[o];
            for (int d = 0; d < HIDDEN; ++d)
                acc += wrow[d] * u_l[k][d];
            inter_l[k][o & 127] = acc;
        }
        __syncthreads();

        // D: squash
        if (tid < NUM_INTEREST) {
            float n2 = 0.0f;
            for (int d = 0; d < HIDDEN; ++d) { float v = inter_l[tid][d]; n2 += v * v; }
            float n = sqrtf(n2);
            scale_l[tid] = (n2 / (1.0f + n2)) / (n + 1e-12f);
        }
        __syncthreads();
        for (int j = 0; j < 4; ++j) {
            int o = tid + j * 256;
            inter_l[o >> 7][o & 127] *= scale_l[o >> 7];
        }
        __syncthreads();

        if (it == ITER_TIMES - 1) break;  // last logit update is dead in the reference

        // E1: v[k,d] = sum_d' W[k*128+d', d] * interest[k,d'] ; beta[k] = bias_k . interest_k
        for (int j = 0; j < 4; ++j) {
            int o = tid + j * 256;
            int k = o >> 7, d = o & 127;
            const float* wk = W_g + (size_t)k * HIDDEN * HIDDEN + d;
            float acc = 0.0f;
            for (int dp = 0; dp < HIDDEN; ++dp)
                acc += wk[(size_t)dp * HIDDEN] * inter_l[k][dp];
            v_bf[k][d] = f2bf(acc);
        }
        if (tid < NUM_INTEREST) {
            float bsum = 0.0f;
            const float* bl = b_lin_g + tid * HIDDEN;
            for (int d = 0; d < HIDDEN; ++d) bsum += bl[d] * inter_l[tid][d];
            beta_l[tid] = bsum;
        }
        __syncthreads();

        // E2: b_logit += item @ v^T + beta  (208x128 @ 128x16pad), WMMA bf16
        //     13 s-tiles round-robined over 8 waves (wave-uniform loop -> EXEC all ones)
        for (int t = wv; t < S_PAD / 16; t += 8) {
            const int stile = t * 16;
            v8f acc = {0.f,0.f,0.f,0.f,0.f,0.f,0.f,0.f};
#pragma unroll
            for (int kd = 0; kd < HIDDEN; kd += 32) {
                v16bf a  = load_frag_A(&item_bf[stile + col][kd], half);
                v16bf bb = load_frag_A(&v_bf[col][kd], half);
                acc = __builtin_amdgcn_wmma_f32_16x16x32_bf16(false, a, false, bb,
                                                              (short)0, acc, false, false);
            }
            if (col < NUM_INTEREST) {
#pragma unroll
                for (int r = 0; r < 8; ++r) {
                    int s = stile + r + rowoff;
                    if (s < SEQ)
                        b_logit[col][s] += acc[r] + beta_l[col];
                }
            }
        }
        __syncthreads();
    }

    // output interest (B, K, d)
    for (int j = 0; j < 4; ++j) {
        int o = tid + j * 256;
        out_g[(size_t)b * (NUM_INTEREST * HIDDEN) + o] = inter_l[o >> 7][o & 127];
    }
}

extern "C" void kernel_launch(void* const* d_in, const int* in_sizes, int n_in,
                              void* d_out, int out_size, void* d_ws, size_t ws_size,
                              hipStream_t stream) {
    const float* item  = (const float*)d_in[0];
    const int*   mask  = (const int*)d_in[1];
    const float* W     = (const float*)d_in[2];
    const float* b_lin = (const float*)d_in[3];
    float* out = (float*)d_out;
    (void)in_sizes; (void)n_in; (void)out_size; (void)d_ws; (void)ws_size;
    capsule_routing_kernel<<<BATCH, 256, 0, stream>>>(item, mask, W, b_lin, out);
}
